// UV_Aggregator_14422500180541
// MI455X (gfx1250) — matrix operand order repaired and checked by reading, not compiled, not run on hardware
//
#include <hip/hip_runtime.h>

typedef __attribute__((ext_vector_type(16))) _Float16 v16h;
typedef __attribute__((ext_vector_type(8)))  _Float16 v8h;
typedef __attribute__((ext_vector_type(8)))  float    v8f;

#define B_    4096
#define H_    200
#define NV_   100000
#define KP1   96     // layer-1 K padded (64 emb + 1 r + zeros)
#define NT    13     // ceil(200/16) row tiles -> 208 padded rows
#define OHP   72     // ohist row stride in halves (padded vs 64)

#define WSW_HALVES 18432                       // weight region in ws (halves)
#define V2E_HALVES ((size_t)NV_ * 64)          // f16 embedding table (halves)

union Frag { v16h f; v8h h[2]; };

// ---------------------------------------------------------------------------
// prep kernel 1: convert weights to f16, padded row-major [N][K] layouts in ws
//   halves [0,     6144) : W1h [64][96]  (col 64 = r-weight, 65..95 = 0)
//   halves [6144, 10240) : W2h [64][64]
//   halves [10240,14336) : A1h [64][64]  (o_history half of att1_w)
//   halves [14336,18432) : A2h [64][64]
// ---------------------------------------------------------------------------
__global__ void prep_weights(const float* __restrict__ w1,
                             const float* __restrict__ w2,
                             const float* __restrict__ a1,
                             const float* __restrict__ a2,
                             _Float16* __restrict__ ws) {
  int i = blockIdx.x * blockDim.x + threadIdx.x;
  if (i < 6144) {
    int n = i / 96, k = i % 96;
    float v = (k < 65) ? w1[n * 65 + k] : 0.f;
    ws[i] = (_Float16)v;
  } else if (i < 10240) {
    int j = i - 6144; int n = j / 64, k = j % 64;
    ws[i] = (_Float16)w2[n * 64 + k];
  } else if (i < 14336) {
    int j = i - 10240; int n = j / 64, k = j % 64;
    ws[i] = (_Float16)a1[n * 128 + k];
  } else if (i < 18432) {
    int j = i - 14336; int n = j / 64, k = j % 64;
    ws[i] = (_Float16)a2[n * 64 + k];
  }
}

// ---------------------------------------------------------------------------
// prep kernel 2: convert the whole v2e table to f16 (halves gather traffic)
// ---------------------------------------------------------------------------
__global__ void prep_v2e(const float* __restrict__ src, _Float16* __restrict__ dst) {
  size_t i = ((size_t)blockIdx.x * blockDim.x + threadIdx.x) * 4;
  if (i < V2E_HALVES) {
    float4 f = *(const float4*)(src + i);
    _Float16* d = dst + i;
    d[0] = (_Float16)f.x; d[1] = (_Float16)f.y;
    d[2] = (_Float16)f.z; d[3] = (_Float16)f.w;
  }
}

// ---------------------------------------------------------------------------
// main kernel: one block per node b; 4 waves, each owns a 16-wide N-subtile
// ---------------------------------------------------------------------------
__launch_bounds__(128)
__global__ void uv_agg_kernel(const int*   __restrict__ nodes,
                              const int*   __restrict__ hist_uv,
                              const float* __restrict__ hist_r,
                              const float* __restrict__ u2e,
                              const float* __restrict__ v2e,
                              const _Float16* __restrict__ v2e_h, // nullptr -> f32 path
                              const float* __restrict__ b1g,
                              const float* __restrict__ b2g,
                              const float* __restrict__ a1wg,    // f32 att1_w (u-fold)
                              const float* __restrict__ ba1g,    // att1_b
                              const float* __restrict__ ba2g,    // att2_b
                              const float* __restrict__ w3g,     // att3_w
                              const _Float16* __restrict__ ws,
                              float* __restrict__ out) {
  __shared__ __align__(16) _Float16 stage[2][16][KP1];
  __shared__ _Float16 ohist[NT * 16][OHP];
  __shared__ float logits[NT * 16];
  __shared__ float redmx[4], redsm[4];

  const int t    = threadIdx.x;
  const int lane = t & 31;
  const int wave = t >> 5;
  const int r    = lane & 15;
  const int hi   = lane >> 4;
  const int n0   = wave * 16;
  const int nn   = n0 + r;          // this lane's output-feature column
  const int b    = blockIdx.x;

  // ---- load this wave's weight B-fragments once (kept in VGPRs) ----
  // B frag from row-major [N][Kp]: lane halves 0..15 = W[nn][kt*32 + hi*16 + 0..15]
  Frag wf1[3], wf2[2], wa1[2], wa2[2];
  {
    const _Float16* W1 = ws;
    const _Float16* W2 = ws + 6144;
    const _Float16* A1 = ws + 10240;
    const _Float16* A2 = ws + 14336;
#pragma unroll
    for (int kt = 0; kt < 3; ++kt) {
      const _Float16* p = W1 + nn * 96 + kt * 32 + hi * 16;
      wf1[kt].h[0] = *(const v8h*)(p);
      wf1[kt].h[1] = *(const v8h*)(p + 8);
    }
#pragma unroll
    for (int kt = 0; kt < 2; ++kt) {
      const _Float16* p = W2 + nn * 64 + kt * 32 + hi * 16;
      wf2[kt].h[0] = *(const v8h*)(p);
      wf2[kt].h[1] = *(const v8h*)(p + 8);
      const _Float16* q = A1 + nn * 64 + kt * 32 + hi * 16;
      wa1[kt].h[0] = *(const v8h*)(q);
      wa1[kt].h[1] = *(const v8h*)(q + 8);
      const _Float16* s = A2 + nn * 64 + kt * 32 + hi * 16;
      wa2[kt].h[0] = *(const v8h*)(s);
      wa2[kt].h[1] = *(const v8h*)(s + 8);
    }
  }

  // ---- per-lane loop-invariant bias registers (no LDS staging) ----
  // att1's u-half contribution is constant over neighbors: fold into bias.
  const float bv1  = b1g[nn];
  const float bv2  = b2g[nn];
  const float bva2 = ba2g[nn];
  const float wv   = w3g[nn];
  float bva1 = ba1g[nn];
  {
    const int node = nodes[b];
    const float* u    = u2e + (size_t)node * 64;
    const float* arow = a1wg + nn * 128 + 64;    // att1_w[nn][64..127]
#pragma unroll 8
    for (int k = 0; k < 64; ++k) bva1 += arow[k] * u[k];
  }

  for (int i = t; i < NT * 16; i += 128) logits[i] = 0.f;
  __syncthreads();

  // A-fragment loader from a 16-row LDS tile, row-major stride KP1
  auto loadA = [&](const _Float16 (*S)[KP1], int kt) -> v16h {
    Frag a;
    a.h[0] = *(const v8h*)&S[r][kt * 32 + hi * 16];
    a.h[1] = *(const v8h*)&S[r][kt * 32 + hi * 16 + 8];
    return a.f;
  };

  // ---- tile loop over neighbor rows ----
  for (int tile = 0; tile < NT; ++tile) {
    // gather 16 rows of v2e embeddings + r into stage[0] (8 threads per row)
    {
      const int row = t >> 3, part = t & 7;
      const int h = tile * 16 + row;
      int idx = 0; float rv = 0.f;
      if (h < H_) { idx = hist_uv[b * H_ + h]; rv = hist_r[b * H_ + h]; }
      if (v2e_h) {
        // f16 table: one b128 load, one b128 LDS store, no conversion
        v8h e = *(const v8h*)(v2e_h + (size_t)idx * 64 + part * 8);
        *(v8h*)&stage[0][row][part * 8] = e;
      } else {
        const float* src = v2e + (size_t)idx * 64 + part * 8;
        float4 f0 = *(const float4*)(src);
        float4 f1 = *(const float4*)(src + 4);
        v8h o;
        o[0] = (_Float16)f0.x; o[1] = (_Float16)f0.y;
        o[2] = (_Float16)f0.z; o[3] = (_Float16)f0.w;
        o[4] = (_Float16)f1.x; o[5] = (_Float16)f1.y;
        o[6] = (_Float16)f1.z; o[7] = (_Float16)f1.w;
        *(v8h*)&stage[0][row][part * 8] = o;
      }
      if (part == 0) {
        v8h z = {};
        *(v8h*)&stage[0][row][64] = z;
        *(v8h*)&stage[0][row][72] = z;
        *(v8h*)&stage[0][row][80] = z;
        *(v8h*)&stage[0][row][88] = z;
        stage[0][row][64] = (_Float16)rv;   // same thread, DS in order
      }
      // prefetch next tile's embedding row into the WGP cache
      // (locality 3 -> WGP scope: fills every cache level on a miss;
      //  SYS scope would stop at GL2, where this data already lives)
      const int h2 = h + 16;
      if (h2 < H_) {
        const int idx2 = hist_uv[b * H_ + h2];
        const void* pf = v2e_h
            ? (const void*)(v2e_h + (size_t)idx2 * 64 + part * 8)
            : (const void*)(v2e   + (size_t)idx2 * 64 + part * 8);
        __builtin_prefetch(pf, 0, 3);
      }
    }
    __syncthreads();

    // Layer 1: [16x96] x [96x16]  -> stage[1]
    {
      v8f c = {bv1, bv1, bv1, bv1, bv1, bv1, bv1, bv1};
#pragma unroll
      for (int kt = 0; kt < 3; ++kt) {
        v16h a = loadA(stage[0], kt);
        c = __builtin_amdgcn_wmma_f32_16x16x32_f16(false, a, false, wf1[kt].f,
                                                   (short)0, c, false, false);
      }
#pragma unroll
      for (int v = 0; v < 8; ++v) {
        float x = c[v] > 0.f ? c[v] : 0.f;
        stage[1][hi * 8 + v][nn] = (_Float16)x;
      }
    }
    __syncthreads();

    // Layer 2: -> o_history (relu), keep f16 copy for att input + final sum
    {
      v8f c = {bv2, bv2, bv2, bv2, bv2, bv2, bv2, bv2};
#pragma unroll
      for (int kt = 0; kt < 2; ++kt) {
        v16h a = loadA(stage[1], kt);
        c = __builtin_amdgcn_wmma_f32_16x16x32_f16(false, a, false, wf2[kt].f,
                                                   (short)0, c, false, false);
      }
#pragma unroll
      for (int v = 0; v < 8; ++v) {
        float x = c[v] > 0.f ? c[v] : 0.f;
        _Float16 xh = (_Float16)x;
        stage[0][hi * 8 + v][nn] = xh;
        ohist[tile * 16 + hi * 8 + v][nn] = xh;
      }
    }
    __syncthreads();

    // ATT1 (K=64, u-part folded into bias) -> stage[1]
    {
      v8f c = {bva1, bva1, bva1, bva1, bva1, bva1, bva1, bva1};
#pragma unroll
      for (int kt = 0; kt < 2; ++kt) {
        v16h a = loadA(stage[0], kt);
        c = __builtin_amdgcn_wmma_f32_16x16x32_f16(false, a, false, wa1[kt].f,
                                                   (short)0, c, false, false);
      }
#pragma unroll
      for (int v = 0; v < 8; ++v) {
        float x = c[v] > 0.f ? c[v] : 0.f;
        stage[1][hi * 8 + v][nn] = (_Float16)x;
      }
    }
    __syncthreads();

    // ATT2 + logit head partials (att3_b dropped: softmax shift-invariant)
    {
      v8f c = {bva2, bva2, bva2, bva2, bva2, bva2, bva2, bva2};
#pragma unroll
      for (int kt = 0; kt < 2; ++kt) {
        v16h a = loadA(stage[1], kt);
        c = __builtin_amdgcn_wmma_f32_16x16x32_f16(false, a, false, wa2[kt].f,
                                                   (short)0, c, false, false);
      }
      float p[8];
#pragma unroll
      for (int v = 0; v < 8; ++v) {
        float x = c[v] > 0.f ? c[v] : 0.f;
        p[v] = x * wv;
      }
#pragma unroll
      for (int off = 1; off <= 8; off <<= 1) {
#pragma unroll
        for (int v = 0; v < 8; ++v) p[v] += __shfl_xor(p[v], off, 32);
      }
      if (r == 0) {
#pragma unroll
        for (int v = 0; v < 8; ++v)
          atomicAdd(&logits[tile * 16 + hi * 8 + v], p[v]);
      }
    }
    // next gather writes stage[0]: safe (all stage[0] reads fenced by the
    // post-ATT1 barrier; ATT2 only reads stage[1], refilled after next barrier)
  }
  __syncthreads();

  // ---- softmax over neighbors (208 padded slots, >=H masked out) ----
  const int h1i = t + 128;
  float l0 = (t   < H_) ? logits[t]   : -3.0e38f;
  float l1 = (h1i < H_) ? logits[h1i] : -3.0e38f;
  float lm = fmaxf(l0, l1);
#pragma unroll
  for (int off = 16; off >= 1; off >>= 1) lm = fmaxf(lm, __shfl_xor(lm, off, 32));
  if (lane == 0) redmx[wave] = lm;
  __syncthreads();
  const float gmax = fmaxf(fmaxf(redmx[0], redmx[1]), fmaxf(redmx[2], redmx[3]));
  float e0 = (t   < H_) ? __expf(l0 - gmax) : 0.f;
  float e1 = (h1i < H_) ? __expf(l1 - gmax) : 0.f;
  float sm = e0 + e1;
#pragma unroll
  for (int off = 16; off >= 1; off >>= 1) sm += __shfl_xor(sm, off, 32);
  if (lane == 0) redsm[wave] = sm;
  __syncthreads();
  logits[t] = e0;                      // unnormalized exp weights
  if (h1i < NT * 16) logits[h1i] = e1;
  const float inv = 1.f / (redsm[0] + redsm[1] + redsm[2] + redsm[3]);
  __syncthreads();

  // ---- attention-weighted neighbor sum: out[b, nn] ----
  float acc = 0.f;
  for (int tile = 0; tile < NT; ++tile) {
#pragma unroll
    for (int v = 0; v < 8; ++v) {
      const int m = tile * 16 + hi * 8 + v;
      acc += logits[m] * (float)ohist[m][nn];
    }
  }
  acc += __shfl_xor(acc, 16, 32);      // combine the two M half-groups (same N)
  if (hi == 0) out[(size_t)b * 64 + nn] = acc * inv;
}

// ---------------------------------------------------------------------------
extern "C" void kernel_launch(void* const* d_in, const int* in_sizes, int n_in,
                              void* d_out, int out_size, void* d_ws, size_t ws_size,
                              hipStream_t stream) {
  const int*   nodes   = (const int*)  d_in[0];
  const int*   hist_uv = (const int*)  d_in[1];
  const float* hist_r  = (const float*)d_in[2];
  const float* u2e     = (const float*)d_in[3];
  const float* v2e     = (const float*)d_in[4];
  const float* w1      = (const float*)d_in[5];
  const float* b1      = (const float*)d_in[6];
  const float* w2      = (const float*)d_in[7];
  const float* b2      = (const float*)d_in[8];
  const float* a1w     = (const float*)d_in[9];
  const float* a1b     = (const float*)d_in[10];
  const float* a2w     = (const float*)d_in[11];
  const float* a2b     = (const float*)d_in[12];
  const float* a3w     = (const float*)d_in[13];
  // d_in[14] (att3_b) intentionally unused: constant logit shift, softmax-invariant

  _Float16* ws  = (_Float16*)d_ws;
  float*    out = (float*)d_out;

  prep_weights<<<(WSW_HALVES + 255) / 256, 256, 0, stream>>>(w1, w2, a1w, a2w, ws);

  // f16 embedding table in scratch if it fits (halves L2 gather traffic)
  const size_t need = (size_t)WSW_HALVES * 2 + V2E_HALVES * 2;
  _Float16* v2e_h = nullptr;
  if (ws_size >= need) {
    v2e_h = ws + WSW_HALVES;
    const size_t nquad = (V2E_HALVES + 3) / 4;
    prep_v2e<<<(int)((nquad + 255) / 256), 256, 0, stream>>>(v2e, v2e_h);
  }

  uv_agg_kernel<<<B_, 128, 0, stream>>>(nodes, hist_uv, hist_r, u2e, v2e, v2e_h,
                                        b1, b2, a1w, a1b, a2b, a3w, ws, out);
}
